// SelfAttention_80625126081172
// MI455X (gfx1250) — matrix operand order repaired and checked by reading, not compiled, not run on hardware
//
#include <hip/hip_runtime.h>
#include <hip/hip_bf16.h>

// ---------------------------------------------------------------------------
// MI455X (gfx1250) fused multi-head self-attention, bf16 WMMA + f32 accum.
//   B=2, S=2048, d_model=512, H=8, depth=64.
// Stage 0: one-time f32->bf16 conversion of x and W (W transposed out-major).
// Stage 1: QKV projection: whole 16x512 x row-tile async-staged to LDS once
//          (ASYNCcnt), then a barrier-free k-loop of v_wmma_f32_16x16x32_bf16.
// Stage 2: flash attention; K/V chunks streamed into double-buffered LDS with
//          global_load_async_to_lds_b128, QK^T + PV on WMMA, f32 online
//          softmax with wave shuffles.
// ---------------------------------------------------------------------------

#define DM     512
#define NH     8
#define DEPTH  64
#define BATCH  2
#define SEQ    2048
#define NEGBIG (-1e9f)

#define XS_STRIDE 520   // 512 + 8 halves pad: keeps 16B alignment, spreads banks

typedef __attribute__((ext_vector_type(16))) __bf16 v16bf;
typedef __attribute__((ext_vector_type(8)))  float  v8f;

union FragBF { v16bf v; unsigned short u[16]; unsigned int w[8]; };

// f32 -> bf16 (round to nearest even)
__device__ inline unsigned short f2bf(float f) {
  unsigned int x = __builtin_bit_cast(unsigned int, f);
  unsigned int r = x + 0x7FFFu + ((x >> 16) & 1u);
  return (unsigned short)(r >> 16);
}

// A-matrix (16-bit, 16x32) per-lane K index for half element hh (contiguous: k = 16*hi + hh)
__device__ inline int a_k(int hh, int hi) {
  return 16 * hi + hh;
}

// 16-byte async global->LDS copy (GVS mode: SGPR base + VGPR byte offset).
// Tracked by ASYNCcnt; completion via s_wait_asynccnt.
__device__ inline void cp_async16(const unsigned short* lds_dst,
                                  const unsigned short* gbase,
                                  unsigned byte_off) {
  unsigned l = (unsigned)(unsigned long long)lds_dst;  // LDS offset in [31:0]
  asm volatile("global_load_async_to_lds_b128 %0, %1, %2"
               :: "v"(l), "v"(byte_off), "s"(gbase) : "memory");
}

// ---------------------------------------------------------------------------
// Stage 0a: x (f32) -> xbf (bf16), 4 elements / thread, packed stores.
// ---------------------------------------------------------------------------
__global__ __launch_bounds__(256)
void cvt_x_kernel(const float* __restrict__ x, unsigned short* __restrict__ xbf) {
  size_t i = ((size_t)blockIdx.x * 256 + threadIdx.x) * 4;
  float4 v = *(const float4*)&x[i];
  unsigned lo = (unsigned)f2bf(v.x) | ((unsigned)f2bf(v.y) << 16);
  unsigned hh = (unsigned)f2bf(v.z) | ((unsigned)f2bf(v.w) << 16);
  *(uint2*)&xbf[i] = make_uint2(lo, hh);
}

// ---------------------------------------------------------------------------
// Stage 0b: W (f32, k-major) -> Wt (bf16, out-major: Wt[mat][n][k]).
// grid=(DM, 3); one k-row per block, coalesced reads, scattered b16 stores.
// ---------------------------------------------------------------------------
__global__ __launch_bounds__(256)
void cvt_w_kernel(const float* __restrict__ Wq, const float* __restrict__ Wk,
                  const float* __restrict__ Wv, unsigned short* __restrict__ Wt) {
  const int k = blockIdx.x, mat = blockIdx.y;
  const float* W = (mat == 0) ? Wq : (mat == 1) ? Wk : Wv;
  unsigned short* dst = Wt + (size_t)mat * DM * DM;
  int n0 = threadIdx.x * 2;
  dst[(size_t)(n0 + 0) * DM + k] = f2bf(W[(size_t)k * DM + n0 + 0]);
  dst[(size_t)(n0 + 1) * DM + k] = f2bf(W[(size_t)k * DM + n0 + 1]);
}

// ---------------------------------------------------------------------------
// Stage 1: QKV projection. grid=(B*S/16, 3*NH); block=128 (4 waves).
// Wave w computes a 16x16 tile at column head*64 + w*16. The 16x512 x-tile is
// async-staged to LDS once; the k-loop then runs with no barriers at all.
// ---------------------------------------------------------------------------
__global__ __launch_bounds__(128)
void qkv_proj_kernel(const unsigned short* __restrict__ xbf,
                     const unsigned short* __restrict__ Wt,
                     const float* __restrict__ bq, const float* __restrict__ bk,
                     const float* __restrict__ bv,
                     unsigned short* __restrict__ Qh,
                     unsigned short* __restrict__ Kh,
                     unsigned short* __restrict__ Vt) {
  __shared__ __align__(16) unsigned short xs[16 * XS_STRIDE];  // 16x512 bf16

  const int t = threadIdx.x;
  const int wv = t >> 5;
  const int lane = t & 31;
  const int ln = lane & 15, hi = lane >> 4;
  const int rowtile = blockIdx.x;
  const int mat = blockIdx.y >> 3;   // 0=Q 1=K 2=V
  const int head = blockIdx.y & 7;
  const int colbase = head * DEPTH + wv * 16;

  const unsigned short* WtMat = Wt + (size_t)mat * DM * DM;
  const float* bias = (mat == 0) ? bq : (mat == 1) ? bk : bv;
  const unsigned short* xrow = xbf + (size_t)rowtile * 16 * DM;

  // async stage of the whole 16x512 bf16 tile: 1024 x 16B, 8 chunks / thread
#pragma unroll
  for (int p = 0; p < 8; ++p) {
    int idx = t + p * 128;            // 0..1023
    int row = idx >> 6;               // 64 x 16B chunks per row
    int c8 = (idx & 63) * 8;          // half index within row
    cp_async16(&xs[row * XS_STRIDE + c8], xrow, (unsigned)((row * DM + c8) * 2));
  }
  asm volatile("s_wait_asynccnt 0x0" ::: "memory");
  __syncthreads();

  v8f acc = {};
  for (int kk = 0; kk < DM; kk += 32) {
    FragBF a, bfr;
#pragma unroll
    for (int j = 0; j < 8; ++j)       // A: k-pairs are contiguous -> b32 loads
      a.w[j] = *(const unsigned int*)&xs[ln * XS_STRIDE + kk + a_k(2 * j, hi)];
    // B 32x16: K contiguous per lane in out-major Wt -> one 32B v16bf load
    bfr.v = *(const v16bf*)&WtMat[(size_t)(colbase + ln) * DM + kk + hi * 16];

    acc = __builtin_amdgcn_wmma_f32_16x16x32_bf16(false, a.v, false, bfr.v,
                                                  (short)0, acc, false, false);
  }

  const float bval = bias[colbase + ln];
#pragma unroll
  for (int r = 0; r < 8; ++r) {
    int m = r + 8 * hi;                 // C layout row
    int g = rowtile * 16 + m;           // flattened (b,s)
    int bb = g >> 11;                   // / SEQ
    int s = g & (SEQ - 1);
    int d = wv * 16 + ln;               // depth index within head
    unsigned short val = f2bf(acc[r] + bval);
    if (mat == 2) {
      Vt[(((head * BATCH + bb) * DEPTH) + d) * SEQ + s] = val;  // depth-major
    } else {
      unsigned short* dst = (mat == 0) ? Qh : Kh;
      dst[(((head * BATCH + bb) * SEQ) + s) * DEPTH + d] = val;
    }
  }
}

// ---------------------------------------------------------------------------
// Stage 2: flash attention. grid=(NH*B, SEQ/128); block=256 (8 waves).
// Double-buffered async K/V staging; wave w owns 16 query rows.
// ---------------------------------------------------------------------------
__global__ __launch_bounds__(256)
void attn_kernel(const unsigned short* __restrict__ Qh,
                 const unsigned short* __restrict__ Kh,
                 const unsigned short* __restrict__ Vt,
                 const float* __restrict__ mask,
                 float* __restrict__ out) {
  __shared__ __align__(16) unsigned short Ks[2][64 * 64];     // [buf][key][d]
  __shared__ __align__(16) unsigned short Vs[2][64 * 64];     // [buf][d][key]
  __shared__ __align__(16) unsigned short Ps[8 * 16 * 64];    // per-wave P

  const int t = threadIdx.x;
  const int wv = t >> 5;
  const int lane = t & 31;
  const int ln = lane & 15, hi = lane >> 4;
  const int hb = blockIdx.x;
  const int h = hb >> 1, b = hb & 1;
  const int q0 = blockIdx.y * 128 + wv * 16;

  const unsigned short* Qbase = Qh + (size_t)(h * BATCH + b) * SEQ * DEPTH;
  const unsigned short* Kbase = Kh + (size_t)(h * BATCH + b) * SEQ * DEPTH;
  const unsigned short* Vbase = Vt + (size_t)(h * BATCH + b) * DEPTH * SEQ;
  const float* mbase = mask + (size_t)b * SEQ * SEQ;

  // issue async copies of one 64-key chunk into LDS buffer `buf`
  auto issue_chunk = [&](int jj, int buf) {
#pragma unroll
    for (int p = 0; p < 2; ++p) {
      int idx = t + p * 256;                       // 0..511 x 16B
      cp_async16(&Ks[buf][idx * 8], Kbase,
                 (unsigned)((jj * DEPTH + idx * 8) * 2));
      int d = idx >> 3, c8 = (idx & 7) << 3;
      cp_async16(&Vs[buf][d * 64 + c8], Vbase,
                 (unsigned)((d * SEQ + jj + c8) * 2));
    }
  };

  issue_chunk(0, 0);   // prefetch first chunk while we load Q fragments

  // Q fragments for the two 32-deep k-steps (k-pairs contiguous -> b32 loads)
  FragBF aQ[2];
#pragma unroll
  for (int ks = 0; ks < 2; ++ks)
#pragma unroll
    for (int j = 0; j < 8; ++j)
      aQ[ks].w[j] = *(const unsigned int*)
          &Qbase[(size_t)(q0 + ln) * DEPTH + ks * 32 + a_k(2 * j, hi)];

  float mrow[8], lrow[8];
  v8f accO[4];
  v8f vzero = {};
#pragma unroll
  for (int r = 0; r < 8; ++r) { mrow[r] = -3.0e38f; lrow[r] = 0.f; }
#pragma unroll
  for (int tt = 0; tt < 4; ++tt) accO[tt] = vzero;

  int cur = 0;
  for (int j = 0; j < SEQ; j += 64) {
    // wait for this wave's async copies, then barrier so all pieces visible;
    // the same barrier guarantees everyone finished reading the other buffer.
    asm volatile("s_wait_asynccnt 0x0" ::: "memory");
    __syncthreads();
    if (j + 64 < SEQ) issue_chunk(j + 64, cur ^ 1);

    // ---- S = Q K^T (4 key subtiles of 16), scale + mask ----
    v8f ST[4];
#pragma unroll
    for (int n = 0; n < 4; ++n) {
      v8f c = vzero;
#pragma unroll
      for (int ks = 0; ks < 2; ++ks) {
        v16bf bK = *(const v16bf*)&Ks[cur][(n * 16 + ln) * 64 + ks * 32 + hi * 16];
        c = __builtin_amdgcn_wmma_f32_16x16x32_bf16(false, aQ[ks].v, false, bK,
                                                    (short)0, c, false, false);
      }
#pragma unroll
      for (int r = 0; r < 8; ++r) {
        int qg = q0 + r + 8 * hi;
        c[r] = c[r] * 0.125f +
               mbase[(size_t)qg * SEQ + j + n * 16 + ln] * NEGBIG;
      }
      ST[n] = c;
    }

    // ---- online softmax over the 64-key chunk ----
#pragma unroll
    for (int r = 0; r < 8; ++r) {
      float tmax = fmaxf(fmaxf(ST[0][r], ST[1][r]), fmaxf(ST[2][r], ST[3][r]));
      tmax = fmaxf(tmax, __shfl_xor(tmax, 1, 16));
      tmax = fmaxf(tmax, __shfl_xor(tmax, 2, 16));
      tmax = fmaxf(tmax, __shfl_xor(tmax, 4, 16));
      tmax = fmaxf(tmax, __shfl_xor(tmax, 8, 16));
      float newm = fmaxf(mrow[r], tmax);
      float corr = __expf(mrow[r] - newm);
      float psum = 0.f;
#pragma unroll
      for (int n = 0; n < 4; ++n) {
        float p = __expf(ST[n][r] - newm);
        psum += p;
        Ps[wv * 1024 + (r + 8 * hi) * 64 + n * 16 + ln] = f2bf(p);
      }
      psum += __shfl_xor(psum, 1, 16);
      psum += __shfl_xor(psum, 2, 16);
      psum += __shfl_xor(psum, 4, 16);
      psum += __shfl_xor(psum, 8, 16);
      lrow[r] = lrow[r] * corr + psum;
      mrow[r] = newm;
#pragma unroll
      for (int tt = 0; tt < 4; ++tt) accO[tt][r] *= corr;
    }

    // per-wave LDS RAW: drain DS counter before reading P back as A-fragments
    asm volatile("s_wait_dscnt 0x0" ::: "memory");

    // ---- O += P V (2 k-steps of 32 keys, 4 depth subtiles) ----
#pragma unroll
    for (int ks2 = 0; ks2 < 2; ++ks2) {
      FragBF aP;
#pragma unroll
      for (int jj = 0; jj < 8; ++jj)
        aP.w[jj] = *(const unsigned int*)
            &Ps[wv * 1024 + ln * 64 + ks2 * 32 + a_k(2 * jj, hi)];
#pragma unroll
      for (int tt = 0; tt < 4; ++tt) {
        v16bf bV = *(const v16bf*)&Vs[cur][(tt * 16 + ln) * 64 + ks2 * 32 + hi * 16];
        accO[tt] = __builtin_amdgcn_wmma_f32_16x16x32_bf16(
            false, aP.v, false, bV, (short)0, accO[tt], false, false);
      }
    }
    cur ^= 1;
  }

  // ---- normalize and write out[b][s][h*64+d] (f32) ----
#pragma unroll
  for (int r = 0; r < 8; ++r) {
    int s = q0 + r + 8 * hi;
    float inv = 1.0f / lrow[r];
#pragma unroll
    for (int tt = 0; tt < 4; ++tt) {
      int d = tt * 16 + ln;
      out[((size_t)b * SEQ + s) * DM + h * DEPTH + d] = accO[tt][r] * inv;
    }
  }
}

// ---------------------------------------------------------------------------
extern "C" void kernel_launch(void* const* d_in, const int* in_sizes, int n_in,
                              void* d_out, int out_size, void* d_ws,
                              size_t ws_size, hipStream_t stream) {
  (void)in_sizes; (void)n_in; (void)out_size; (void)ws_size;
  const float* x    = (const float*)d_in[0];
  const float* mask = (const float*)d_in[1];
  const float* Wq   = (const float*)d_in[2];
  const float* bq   = (const float*)d_in[3];
  const float* Wk   = (const float*)d_in[4];
  const float* bk   = (const float*)d_in[5];
  const float* Wv   = (const float*)d_in[6];
  const float* bv   = (const float*)d_in[7];

  const size_t nX   = (size_t)BATCH * SEQ * DM;       // 2,097,152
  const size_t nW   = (size_t)3 * DM * DM;            //   786,432
  const size_t nMat = (size_t)NH * BATCH * SEQ * DEPTH;

  unsigned short* xbf = (unsigned short*)d_ws;
  unsigned short* Wt  = xbf + nX;
  unsigned short* Qh  = Wt + nW;
  unsigned short* Kh  = Qh + nMat;
  unsigned short* Vt  = Kh + nMat;
  float* out = (float*)d_out;

  cvt_x_kernel<<<dim3(nX / (256 * 4)), 256, 0, stream>>>(x, xbf);
  cvt_w_kernel<<<dim3(DM, 3), 256, 0, stream>>>(Wq, Wk, Wv, Wt);
  qkv_proj_kernel<<<dim3((BATCH * SEQ) / 16, 3 * NH), 128, 0, stream>>>(
      xbf, Wt, bq, bk, bv, Qh, Kh, Vt);
  attn_kernel<<<dim3(NH * BATCH, SEQ / 128), 256, 0, stream>>>(
      Qh, Kh, Vt, mask, out);
}
